// NCDEExposure_66743791780172
// MI455X (gfx1250) — matrix operand order repaired and checked by reading, not compiled, not run on hardware
//
#include <hip/hip_runtime.h>
#include <hip/hip_bf16.h>

typedef __attribute__((ext_vector_type(16))) _Float16 v16h;
typedef __attribute__((ext_vector_type(8)))  float    v8f;

// Problem dims
#define Bn 128
#define Tn 512
#define Cn 64
#define Hn 128
#define Sn 64
#define Fn 8
#define Wd 32   // WIDTH

// A-fragment (16-bit A, 16x32) K index, CDNA5 ISA 7.12.2:
// lanes 0-15 hold K 0-7 (e<8) and 16-23 (e>=8); lanes 16-31 hold K 8-15 and 24-31.
__device__ __forceinline__ int a_kidx(int e, int lane) {
    int half = lane >> 4;
    return (e < 8) ? (half * 8 + e) : (16 + half * 8 + (e - 8));
}
// Inverse map: for within-chunk K (0..31) and row b (0..15), the (lane, element)
// slot inside the A fragment:
__device__ __forceinline__ int a_slot(int b, int k) {     // returns lane*16+e
    int laneHi = (k >> 3) & 1;
    int e = (k & 7) + ((k >> 4) << 3);
    return (laneHi * 16 + b) * 16 + e;
}
// B-fragment K index within a 32-K chunk (same order as A): lanes 0-15 -> K 0-7,16-23; lanes 16-31 -> K 8-15,24-31
__device__ __forceinline__ int b_kidx(int e, int lane) {
    return a_kidx(e, lane);
}

// Branch-free tanh: (e^{2x}-1)/(e^{2x}+1) with clamp; v_exp_f32 + v_rcp_f32.
__device__ __forceinline__ float tanh_fast(float x) {
    float xc = fminf(fmaxf(x, -9.f), 9.f);
    float t = __builtin_amdgcn_exp2f(xc * 2.8853900817779268f);  // exp(2*xc)
    return (t - 1.f) * __builtin_amdgcn_rcpf(t + 1.f);
}

// ---------------------------------------------------------------------------
// Weight pre-swizzle to f16 WMMA B-fragment layout:
//   flat index = ((chunk*NT + ntile)*32 + lane)*16 + e ; value = W[n][k]
// B3 special: chunk = c (0..63), K-in-chunk = w (0..31), value = W3[(h*64+c)*32+w]
// ---------------------------------------------------------------------------
__global__ void prep_weights(const float* __restrict__ W1, const float* __restrict__ W2,
                             const float* __restrict__ W3, const float* __restrict__ b3,
                             const float* __restrict__ Wro,
                             _Float16* __restrict__ B1, _Float16* __restrict__ B2,
                             _Float16* __restrict__ B3, _Float16* __restrict__ Bb3,
                             _Float16* __restrict__ Bro) {
    int gid = blockIdx.x * blockDim.x + threadIdx.x;  // 0 .. 262143
    int e = gid & 15, lane = (gid >> 4) & 31, t = gid >> 9;
    int nl = lane & 15;
    int kin = b_kidx(e, lane);

    if (gid < 4096) {               // B1: 4 chunks x 2 ntiles  (W1: 32x128)
        int nt = t % 2, kc = t / 2;
        B1[gid] = (_Float16)W1[(nt * 16 + nl) * 128 + kc * 32 + kin];
    }
    if (gid < 1024) {               // B2: 1 chunk x 2 ntiles   (W2: 32x32)
        int nt = t;
        B2[gid] = (_Float16)W2[(nt * 16 + nl) * 32 + kin];
    }
    if (gid < 8192) {               // Bb3: 2 chunks x 8 ntiles (b3 reshaped H x C)
        int nt = t & 7, kc = t >> 3;
        int h = nt * 16 + nl, c = kc * 32 + kin;
        Bb3[gid] = (_Float16)b3[h * 64 + c];
    }
    if (gid < 65536) {              // Bro: 4 chunks x 32 ntiles (W_ro: 512x128)
        int nt = t & 31, kc = t >> 5;
        Bro[gid] = (_Float16)Wro[(nt * 16 + nl) * 128 + kc * 32 + kin];
    }
    if (gid < 262144) {             // B3: 64 c-chunks x 8 ntiles (W3: 8192x32)
        int nt = t & 7, c = t >> 3;
        int h = nt * 16 + nl, w = kin;
        B3[gid] = (_Float16)W3[(h * 64 + c) * 32 + w];
    }
}

// ---------------------------------------------------------------------------
// y0 = nan0(cov[:,0]) @ W_init^T + b_init  -> hidden[:, 0, :]
// ---------------------------------------------------------------------------
__global__ void init_y0(const float* __restrict__ cov, const float* __restrict__ Wi,
                        const float* __restrict__ bi, float* __restrict__ hidden) {
    int o = blockIdx.x * blockDim.x + threadIdx.x;  // 128*128
    int b = o >> 7, h = o & 127;
    const float* cv = cov + (size_t)b * Tn * Cn;
    float s = bi[h];
    for (int c = 0; c < Cn; ++c) {
        float x = cv[c]; x = (x != x) ? 0.f : x;
        s += x * Wi[h * Cn + c];
    }
    hidden[((size_t)b * Tn) * Hn + h] = s;
}

// ---------------------------------------------------------------------------
// Persistent DOPRI scan: one workgroup per 16 batches, 8 waves.
// All WMMA A operands are staged in LDS in f16 A-fragment order, so each
// fragment is a single 32B ds read per lane.
// ---------------------------------------------------------------------------
__global__ __launch_bounds__(256, 1) void scan_kernel(
    const float* __restrict__ times, const float* __restrict__ cov,
    const v16h* __restrict__ B1v, const v16h* __restrict__ B2v,
    const v16h* __restrict__ B3v, const v16h* __restrict__ Bb3v,
    const float* __restrict__ b1g, const float* __restrict__ b2g,
    float* __restrict__ hidden) {
    __shared__ float yL[2048];                     // 16 x 128 state (f32)
    __shared__ float kL[5][2048];                  // k1..k5 (k6 reuses slot 1)
    __shared__ __align__(16) float dxsL[1024];     // 16 x 64 dx (f32, for scales)
    __shared__ v16h ysA[128];                      // stage input, A-frag layout (4 chunks)
    __shared__ v16h dxA[64];                       // dx, A-frag layout (2 chunks)
    __shared__ v16h h1A[32];                       // h1, A-frag layout (1 chunk)
    __shared__ v16h h2A[32];                       // h2, A-frag layout (1 chunk)
    _Float16* ysAh = (_Float16*)ysA;
    _Float16* dxAh = (_Float16*)dxA;
    _Float16* h1Ah = (_Float16*)h1A;
    _Float16* h2Ah = (_Float16*)h2A;

    const int tid = threadIdx.x, lane = tid & 31, wave = tid >> 5;
    const int b0 = blockIdx.x * 16;
    const int br = lane & 15;

    for (int i = tid; i < 2048; i += 256) {
        int bl = i >> 7, h = i & 127;
        yL[i] = hidden[((size_t)(b0 + bl) * Tn) * Hn + h];
    }
    __syncthreads();

    for (int t = 0; t < Tn - 1; ++t) {
        float dt = times[t + 1] - times[t];
        // cubic Hermite coefficients, 4 (b,c) items per thread, kept in regs
        float cbR[4], ccR[4], cdR[4];
        for (int u = 0; u < 4; ++u) {
            int idx = tid * 4 + u; int bl = idx >> 6, c = idx & 63;
            const float* cv = cov + ((size_t)(b0 + bl) * Tn + t) * Cn + c;
            float y0v = cv[0];  y0v = (y0v != y0v) ? 0.f : y0v;
            float y1v = cv[Cn]; y1v = (y1v != y1v) ? 0.f : y1v;
            float v1 = (y1v - y0v) / dt;
            float v0 = v1;
            if (t > 0) {
                float ym = cv[-Cn]; ym = (ym != ym) ? 0.f : ym;
                float dtp = times[t] - times[t - 1];
                v0 = (y0v - ym) / dtp;
            }
            cbR[u] = v0;
            ccR[u] = (3.f * (y1v - y0v) / dt - 2.f * v0 - v1) / dt;
            cdR[u] = (2.f * (y0v - y1v) / dt + v0 + v1) / (dt * dt);
        }

        for (int j = 0; j < 6; ++j) {
            // ---- stage setup: dx (f32 + A-layout f16) and ys (A-layout f16) ----
            float al;
            switch (j) {
                case 0: al = 0.f; break;      case 1: al = 0.2f; break;
                case 2: al = 0.3f; break;     case 3: al = 0.8f; break;
                case 4: al = 8.f / 9.f; break; default: al = 1.f; break;
            }
            float s = al * dt;
            for (int u = 0; u < 4; ++u) {
                int idx = tid * 4 + u; int bl = idx >> 6, c = idx & 63;
                float dx = cbR[u] + 2.f * ccR[u] * s + 3.f * cdR[u] * (s * s);
                dxsL[idx] = dx;
                dxAh[(c >> 5) * 512 + a_slot(bl, c & 31)] = (_Float16)dx;
            }
            {
                int i0 = tid * 8;
                for (int u = 0; u < 8; ++u) {
                    int i = i0 + u; float a = 0.f;
                    float ysv;
                    switch (j) {
                        case 0: ysv = yL[i]; break;
                        case 1: a = 0.2f * kL[0][i]; ysv = yL[i] + dt * a; break;
                        case 2: a = (3.f/40.f)*kL[0][i] + (9.f/40.f)*kL[1][i];
                                ysv = yL[i] + dt * a; break;
                        case 3: a = (44.f/45.f)*kL[0][i] + (-56.f/15.f)*kL[1][i] + (32.f/9.f)*kL[2][i];
                                ysv = yL[i] + dt * a; break;
                        case 4: a = (19372.f/6561.f)*kL[0][i] + (-25360.f/2187.f)*kL[1][i]
                                  + (64448.f/6561.f)*kL[2][i] + (-212.f/729.f)*kL[3][i];
                                ysv = yL[i] + dt * a; break;
                        default: a = (9017.f/3168.f)*kL[0][i] + (-355.f/33.f)*kL[1][i]
                                   + (46732.f/5247.f)*kL[2][i] + (49.f/176.f)*kL[3][i]
                                   + (-5103.f/18656.f)*kL[4][i];
                                 ysv = yL[i] + dt * a; break;
                    }
                    int bl = i >> 7, h = i & 127;
                    ysAh[(h >> 5) * 512 + a_slot(bl, h & 31)] = (_Float16)ysv;
                }
            }
            __syncthreads();

            // ---- GEMM1: (16x128)x(128->32), waves 0-1, 4 K-chunks ----
            if (wave < 2) {
                int nt = wave;
                v8f acc = {};
                for (int kc = 0; kc < 4; ++kc) {
                    acc = __builtin_amdgcn_wmma_f32_16x16x32_f16(
                        false, ysA[kc * 32 + lane], false,
                        B1v[(kc * 2 + nt) * 32 + lane], (short)0, acc, false, false);
                }
                int n = nt * 16 + br;
                float bias = b1g[n];
                for (int r = 0; r < 8; ++r) {
                    int m = r + ((lane >> 4) << 3);
                    h1Ah[a_slot(m, n)] = (_Float16)tanh_fast(acc[r] + bias);
                }
            }
            __syncthreads();

            // ---- GEMM2: (16x32)x(32->32), waves 0-1 ----
            if (wave < 2) {
                int nt = wave;
                v8f acc = {};
                acc = __builtin_amdgcn_wmma_f32_16x16x32_f16(
                    false, h1A[lane], false, B2v[nt * 32 + lane], (short)0, acc, false, false);
                int n = nt * 16 + br;
                float bias = b2g[n];
                for (int r = 0; r < 8; ++r) {
                    int m = r + ((lane >> 4) << 3);
                    h2Ah[a_slot(m, n)] = (_Float16)tanh_fast(acc[r] + bias);
                }
            }
            __syncthreads();

            // ---- GEMM3: k = dx@b3r + Sum_c (h2 * dx[:,c]) @ W3chunk(c); 8 waves ----
            {
                int nt = wave;
                v16h ha = h2A[lane];
                v8f acc = {};
                for (int kc = 0; kc < 2; ++kc) {   // bias projection GEMM
                    acc = __builtin_amdgcn_wmma_f32_16x16x32_f16(
                        false, dxA[kc * 32 + lane], false,
                        Bb3v[(kc * 8 + nt) * 32 + lane], (short)0, acc, false, false);
                }
                const float4* dx4 = (const float4*)(dxsL + br * 64);
                for (int c4 = 0; c4 < 16; ++c4) {  // main 32->8192 GEMM, dx as A-scale
                    if (c4 + 1 < 16)
                        __builtin_prefetch((const void*)&B3v[(((c4 + 1) * 4) * 8 + nt) * 32 + lane], 0, 0);
                    float4 d = dx4[c4];
                    float dv[4] = {d.x, d.y, d.z, d.w};
#pragma unroll
                    for (int u = 0; u < 4; ++u) {
                        int c = c4 * 4 + u;
                        v16h a2 = ha * (_Float16)dv[u];
                        acc = __builtin_amdgcn_wmma_f32_16x16x32_f16(
                            false, a2, false, B3v[(c * 8 + nt) * 32 + lane], (short)0, acc, false, false);
                    }
                }
                int slot = (j == 5) ? 1 : j;       // k6 overwrites k2 (unused in combine)
                int n = nt * 16 + br;
                for (int r = 0; r < 8; ++r) {
                    int m = r + ((lane >> 4) << 3);
                    kL[slot][m * 128 + n] = acc[r];
                }
            }
            __syncthreads();
        }

        // ---- 5th-order combine + emit hidden state ----
        {
            int i0 = tid * 8;
            for (int u = 0; u < 8; ++u) {
                int i = i0 + u;
                float yn = yL[i] + dt * ((35.f/384.f)  * kL[0][i] + (500.f/1113.f) * kL[2][i]
                                       + (125.f/192.f) * kL[3][i] + (-2187.f/6784.f)* kL[4][i]
                                       + (11.f/84.f)   * kL[1][i]);
                yL[i] = yn;
                int bl = i >> 7, h = i & 127;
                hidden[((size_t)(b0 + bl) * Tn + (t + 1)) * Hn + h] = yn;
            }
        }
        __syncthreads();
    }
}

// ---------------------------------------------------------------------------
// Readout: out(65536x512) = hidden(65536x128) @ W_ro^T + b_ro, WMMA tiled.
// block = 16 rows (staged to LDS in f16 A-frag layout); 8 waves x 4 grid-y
// cover the 32 N-tiles.
// ---------------------------------------------------------------------------
__global__ __launch_bounds__(256, 2) void readout_kernel(
    const float* __restrict__ hidden, const v16h* __restrict__ Brov,
    const float* __restrict__ bro, float* __restrict__ out) {
    __shared__ v16h AtA[128];                      // 4 chunks x 32 lanes
    _Float16* AtAh = (_Float16*)AtA;
    const int tid = threadIdx.x, lane = tid & 31, wave = tid >> 5;
    const size_t row0 = (size_t)blockIdx.x * 16;
    for (int i = tid; i < 2048; i += 256) {
        int m = i >> 7, h = i & 127;
        AtAh[(h >> 5) * 512 + a_slot(m, h & 31)] = (_Float16)hidden[row0 * 128 + i];
    }
    __syncthreads();

    int nt = blockIdx.y * 8 + wave;
    int br = lane & 15;
    v8f acc = {};
    for (int kc = 0; kc < 4; ++kc) {
        acc = __builtin_amdgcn_wmma_f32_16x16x32_f16(
            false, AtA[kc * 32 + lane], false,
            Brov[(kc * 32 + nt) * 32 + lane], (short)0, acc, false, false);
    }
    int n = nt * 16 + br;
    float bias = bro[n];
    for (int r = 0; r < 8; ++r) {
        int m = r + ((lane >> 4) << 3);
        out[(row0 + m) * 512 + n] = acc[r] + bias;
    }
}

// ---------------------------------------------------------------------------
extern "C" void kernel_launch(void* const* d_in, const int* in_sizes, int n_in,
                              void* d_out, int out_size, void* d_ws, size_t ws_size,
                              hipStream_t stream) {
    const float* times = (const float*)d_in[0];
    const float* cov   = (const float*)d_in[1];
    const float* Winit = (const float*)d_in[2];
    const float* binit = (const float*)d_in[3];
    const float* W1    = (const float*)d_in[4];
    const float* b1    = (const float*)d_in[5];
    const float* W2    = (const float*)d_in[6];
    const float* b2    = (const float*)d_in[7];
    const float* W3    = (const float*)d_in[8];
    const float* b3    = (const float*)d_in[9];
    const float* Wro   = (const float*)d_in[10];
    const float* bro   = (const float*)d_in[11];

    char* wsb = (char*)d_ws;
    constexpr size_t HID_B = (size_t)Bn * Tn * Hn * 4;          // 33,554,432
    float*    hidden = (float*)wsb;
    _Float16* B1h  = (_Float16*)(wsb + HID_B);                  //   8,192 B
    _Float16* B2h  = (_Float16*)(wsb + HID_B + 8192);           //   2,048 B
    _Float16* Bb3h = (_Float16*)(wsb + HID_B + 8192 + 2048);    //  16,384 B
    _Float16* Broh = (_Float16*)(wsb + HID_B + 8192 + 2048 + 16384);            // 131,072 B
    _Float16* B3h  = (_Float16*)(wsb + HID_B + 8192 + 2048 + 16384 + 131072);   // 524,288 B

    prep_weights<<<262144 / 256, 256, 0, stream>>>(W1, W2, W3, b3, Wro,
                                                   B1h, B2h, B3h, Bb3h, Broh);
    init_y0<<<(Bn * Hn) / 256, 256, 0, stream>>>(cov, Winit, binit, hidden);
    scan_kernel<<<Bn / 16, 256, 0, stream>>>(times, cov,
                                             (const v16h*)B1h, (const v16h*)B2h,
                                             (const v16h*)B3h, (const v16h*)Bb3h,
                                             b1, b2, hidden);
    readout_kernel<<<dim3((Bn * Tn) / 16, 4), 256, 0, stream>>>(
        hidden, (const v16h*)Broh, bro, (float*)d_out);
}
